// AttentionalPropagation_62534723829922
// MI455X (gfx1250) — compile-verified, matrix operand order checked by reading, not run on hardware
//
#include <hip/hip_runtime.h>

typedef __bf16 bf16_t;
typedef __attribute__((ext_vector_type(16))) __bf16 bv16;
typedef __attribute__((ext_vector_type(8)))  __bf16 bv8;
typedef __attribute__((ext_vector_type(8)))  float  fv8;

#define WMMA_BF16(A, Bm, Cm) \
  __builtin_amdgcn_wmma_f32_16x16x32_bf16(false, (A), false, (Bm), (short)0, (Cm), false, false)

static __device__ __forceinline__ unsigned pk2(float a, float b) {
  unsigned short lo = __builtin_bit_cast(unsigned short, (__bf16)a);
  unsigned short hi = __builtin_bit_cast(unsigned short, (__bf16)b);
  return (unsigned)lo | ((unsigned)hi << 16);
}

// x: [B=64, C=512, L=4096] f32.  One block handles 16 consecutive l's, one wave per l.
__launch_bounds__(512, 1)
__global__ void attnprop_kernel(const float* __restrict__ x, float* __restrict__ out) {
  constexpr int Bn = 64, Cn = 512, Ln = 4096, LT = 16, CH = 32;
  // Phase 1: qsm = [l][b(64)][c(32)] bf16.  Phase 3: qsm = [l][c(32)][b(64)] bf16. 64KB.
  __shared__ alignas(16) bf16_t qsm[LT * 64 * CH];
  // Output staging: [l][c_half(16)][n(64)] f32. 64KB.
  __shared__ alignas(16) float msgs[LT * 16 * 64];

  const int tid  = threadIdx.x;
  const int w    = tid >> 5;      // wave id == local l
  const int lane = tid & 31;
  const int h    = lane >> 4;     // lane half (WMMA layout)
  const int ln   = lane & 15;
  const int l0   = blockIdx.x * LT;
  unsigned* qs32 = (unsigned*)qsm;
  const float softscale = 0.044194173824159216f; // 1/sqrt(512)

  // ---------------- Phase 1: T[j][i] = S^T tiles, S = Q Q^T ----------------
  fv8 acc[4][4];
  #pragma unroll
  for (int j = 0; j < 4; ++j)
    #pragma unroll
    for (int i = 0; i < 4; ++i)
      #pragma unroll
      for (int r = 0; r < 8; ++r) acc[j][i][r] = 0.0f;

  for (int cc = 0; cc < Cn; cc += CH) {
    __syncthreads();
    // cooperative load: each thread owns (b, c-pair), 16 l's contiguous (64B runs)
    #pragma unroll
    for (int rep = 0; rep < 2; ++rep) {
      int pi = rep * 512 + tid;
      int b  = pi >> 4;
      int cp = pi & 15;
      const float4* p0 = (const float4*)(x + ((size_t)b * Cn + (cc + 2 * cp)) * Ln + l0);
      const float4* p1 = (const float4*)((const float*)p0 + Ln);
      float4 a0 = p0[0], a1 = p0[1], a2 = p0[2], a3 = p0[3];
      float4 b0 = p1[0], b1 = p1[1], b2 = p1[2], b3 = p1[3];
      float va[16] = {a0.x,a0.y,a0.z,a0.w, a1.x,a1.y,a1.z,a1.w,
                      a2.x,a2.y,a2.z,a2.w, a3.x,a3.y,a3.z,a3.w};
      float vb[16] = {b0.x,b0.y,b0.z,b0.w, b1.x,b1.y,b1.z,b1.w,
                      b2.x,b2.y,b2.z,b2.w, b3.x,b3.y,b3.z,b3.w};
      #pragma unroll
      for (int l = 0; l < 16; ++l)
        qs32[(l * 64 + b) * 16 + cp] = pk2(va[l], vb[l]);
    }
    __syncthreads();

    // A fragments (16x32 bf16) for 4 row-blocks; B frag of block m == A frag of block m.
    bv16 fr[4];
    #pragma unroll
    for (int i = 0; i < 4; ++i) {
      const bv8* base = (const bv8*)(qsm + (size_t)(w * 64 + i * 16 + ln) * CH);
      bv8 lo = base[h];       // K = 8h .. 8h+7
      bv8 hi = base[2 + h];   // K = 16+8h .. 16+8h+7
      #pragma unroll
      for (int t = 0; t < 8; ++t) { fr[i][t] = lo[t]; fr[i][8 + t] = hi[t]; }
    }
    #pragma unroll
    for (int j = 0; j < 4; ++j)
      #pragma unroll
      for (int i = 0; i < 4; ++i)
        acc[j][i] = WMMA_BF16(fr[j], fr[i], acc[j][i]);   // T tile (j,i) = S[m=16j+M][n=16i+N]
  }

  // ---------------- Phase 2: softmax over m (per column n), pack P^T B-fragments ----
  // Lane holds, per i: n = 16i+ln fixed; m = 16j + 8h + r  (32 values; other 32 in lane^16)
  bv16 pf[2][4];
  #pragma unroll
  for (int i = 0; i < 4; ++i) {
    float mx = -3.0e38f;
    #pragma unroll
    for (int j = 0; j < 4; ++j)
      #pragma unroll
      for (int r = 0; r < 8; ++r) mx = fmaxf(mx, acc[j][i][r]);
    mx = fmaxf(mx, __shfl_xor(mx, 16, 32));
    float sum = 0.0f;
    #pragma unroll
    for (int j = 0; j < 4; ++j)
      #pragma unroll
      for (int r = 0; r < 8; ++r) {
        float p = __expf((acc[j][i][r] - mx) * softscale);
        acc[j][i][r] = p;
        sum += p;
      }
    sum += __shfl_xor(sum, 16, 32);
    float rs = 1.0f / sum;
    // B-operand fragment of P^T for k-chunk kc: K = kc*32 + (v>=4)*16 + 8h + 2(v&3)+t
    #pragma unroll
    for (int kc = 0; kc < 2; ++kc)
      #pragma unroll
      for (int v = 0; v < 8; ++v) {
        int j  = 2 * kc + (v >> 2);
        int r0 = 2 * (v & 3);
        pf[kc][i][2 * v]     = (__bf16)(acc[j][i][r0]     * rs);
        pf[kc][i][2 * v + 1] = (__bf16)(acc[j][i][r0 + 1] * rs);
      }
  }

  // ---------------- Phase 3: msg^T = Q^T * P^T, residual add, store ----------------
  for (int cc = 0; cc < Cn; cc += CH) {
    __syncthreads();
    // cooperative load, staged TRANSPOSED: qsm = [l][c][b], thread owns (c, b-pair)
    #pragma unroll
    for (int rep = 0; rep < 2; ++rep) {
      int pi = rep * 512 + tid;
      int c  = pi >> 5;
      int b  = (pi & 31) * 2;
      const float4* p0 = (const float4*)(x + ((size_t)b * Cn + cc + c) * Ln + l0);
      const float4* p1 = (const float4*)(x + ((size_t)(b + 1) * Cn + cc + c) * Ln + l0);
      float4 a0 = p0[0], a1 = p0[1], a2 = p0[2], a3 = p0[3];
      float4 b0 = p1[0], b1 = p1[1], b2 = p1[2], b3 = p1[3];
      float va[16] = {a0.x,a0.y,a0.z,a0.w, a1.x,a1.y,a1.z,a1.w,
                      a2.x,a2.y,a2.z,a2.w, a3.x,a3.y,a3.z,a3.w};
      float vb[16] = {b0.x,b0.y,b0.z,b0.w, b1.x,b1.y,b1.z,b1.w,
                      b2.x,b2.y,b2.z,b2.w, b3.x,b3.y,b3.z,b3.w};
      #pragma unroll
      for (int l = 0; l < 16; ++l)
        qs32[(l * CH + c) * 32 + (b >> 1)] = pk2(va[l], vb[l]);
    }
    __syncthreads();

    // D tiles: msg^T[c_local 32][n 64] accumulated over 2 K-chunks of m
    fv8 d[2][4];
    #pragma unroll
    for (int ct = 0; ct < 2; ++ct)
      #pragma unroll
      for (int i = 0; i < 4; ++i)
        #pragma unroll
        for (int r = 0; r < 8; ++r) d[ct][i][r] = 0.0f;

    #pragma unroll
    for (int kc = 0; kc < 2; ++kc) {
      bv16 af[2];
      #pragma unroll
      for (int ct = 0; ct < 2; ++ct) {
        const bv8* base = (const bv8*)(qsm + (size_t)(w * CH + ct * 16 + ln) * 64);
        bv8 lo = base[kc * 4 + h];       // m = kc*32 + 8h ..
        bv8 hi = base[kc * 4 + 2 + h];   // m = kc*32 + 16 + 8h ..
        #pragma unroll
        for (int t = 0; t < 8; ++t) { af[ct][t] = lo[t]; af[ct][8 + t] = hi[t]; }
      }
      #pragma unroll
      for (int ct = 0; ct < 2; ++ct)
        #pragma unroll
        for (int i = 0; i < 4; ++i)
          d[ct][i] = WMMA_BF16(af[ct], pf[kc][i], d[ct][i]);
    }

    // stage + coalesced residual write, one 16-c half at a time
    const size_t rowStride = (size_t)Cn * Ln;
    #pragma unroll
    for (int ct = 0; ct < 2; ++ct) {
      #pragma unroll
      for (int i = 0; i < 4; ++i)
        #pragma unroll
        for (int r = 0; r < 8; ++r)
          msgs[(w * 16 + (r + 8 * h)) * 64 + (i * 16 + ln)] = d[ct][i][r];
      __syncthreads();
      {
        int pi = ct * 512 + tid;
        int c  = pi >> 5;            // in [ct*16, ct*16+16)
        int b  = (pi & 31) * 2;
        int cl = c & 15;
        const float* px = x   + ((size_t)b * Cn + cc + c) * Ln + l0;
        float*       po = out + ((size_t)b * Cn + cc + c) * Ln + l0;
        #pragma unroll
        for (int lq = 0; lq < 4; ++lq) {
          float4 xv0 = ((const float4*)px)[lq];
          float4 xv1 = ((const float4*)(px + rowStride))[lq];
          float4 o0, o1;
          o0.x = xv0.x + msgs[((4 * lq + 0) * 16 + cl) * 64 + b];
          o0.y = xv0.y + msgs[((4 * lq + 1) * 16 + cl) * 64 + b];
          o0.z = xv0.z + msgs[((4 * lq + 2) * 16 + cl) * 64 + b];
          o0.w = xv0.w + msgs[((4 * lq + 3) * 16 + cl) * 64 + b];
          o1.x = xv1.x + msgs[((4 * lq + 0) * 16 + cl) * 64 + b + 1];
          o1.y = xv1.y + msgs[((4 * lq + 1) * 16 + cl) * 64 + b + 1];
          o1.z = xv1.z + msgs[((4 * lq + 2) * 16 + cl) * 64 + b + 1];
          o1.w = xv1.w + msgs[((4 * lq + 3) * 16 + cl) * 64 + b + 1];
          ((float4*)po)[lq] = o0;
          ((float4*)(po + rowStride))[lq] = o1;
        }
      }
      __syncthreads();
    }
  }
}

extern "C" void kernel_launch(void* const* d_in, const int* in_sizes, int n_in,
                              void* d_out, int out_size, void* d_ws, size_t ws_size,
                              hipStream_t stream) {
  const float* x = (const float*)d_in[0];
  float* out = (float*)d_out;
  (void)in_sizes; (void)n_in; (void)out_size; (void)d_ws; (void)ws_size;
  dim3 grid(4096 / 16);   // 256 workgroups, one per 16 consecutive l
  dim3 block(512);        // 16 waves (wave32), one wave per l
  hipLaunchKernelGGL(attnprop_kernel, grid, block, 0, stream, x, out);
}
